// Llama4StyleExperts_71640054497665
// MI455X (gfx1250) — matrix operand order repaired and checked by reading, not compiled, not run on hardware
//
#include <hip/hip_runtime.h>

// MoE experts forward (Llama4-style): per expert
//   gate_up = h @ Wgu ; act = up * silu(gate) ; out = act @ Wd
// bf16 WMMA GEMMs (v_wmma_f32_16x16x32_bf16), fp32->bf16 conversion fused into
// LDS staging (v_cvt_pk_bf16_f32 when available), SwiGLU fused into GEMM1
// epilogue, bf16 act tile in d_ws feeding GEMM2 directly. GEMM2 A-tile staging
// uses CDNA5 async global->LDS copies (ASYNCcnt) when the builtin exists.
// Tiles 128x128x64, 8 waves/WG (wave32), wave = 64x32 out via 4x2 WMMA accums.

#define NEXP   8
#define HID    2048
#define FFN    4096
#define TOK    1024

#define BM 128
#define BN 128
#define BK 64
#define LDA 72   // halves: row stride of LDS A tile  [BM][LDA]
#define LDB 72   // halves: row stride of LDS B tile  [BN][LDB] (transposed)

typedef __attribute__((ext_vector_type(16))) __bf16 v16bf;
typedef __attribute__((ext_vector_type(2)))  __bf16 v2bf;
typedef __attribute__((ext_vector_type(8)))  float  v8f;
typedef __attribute__((ext_vector_type(4)))  float  f4x;
typedef __attribute__((ext_vector_type(4)))  int    i4x;
typedef __attribute__((ext_vector_type(2)))  unsigned int u2x;

#if __has_builtin(__builtin_amdgcn_global_load_async_to_lds_b128)
typedef __attribute__((address_space(1))) i4x* gi4p;   // global-AS vector ptr
typedef __attribute__((address_space(3))) i4x* li4p;   // LDS-AS vector ptr
#endif

// fp32 -> bf16 bits, round-to-nearest-even (pure VALU fallback)
static __device__ __forceinline__ unsigned short f2bf(float f) {
  unsigned int u = __float_as_uint(f);
  u += 0x7fffu + ((u >> 16) & 1u);
  return (unsigned short)(u >> 16);
}

// pack two fp32 -> packed 2xbf16 in one dword (v_cvt_pk_bf16_f32 if present)
static __device__ __forceinline__ unsigned int pack2bf(float x, float y) {
#if __has_builtin(__builtin_amdgcn_cvt_pk_bf16_f32)
  union { v2bf v; unsigned int u; } c;
  c.v = __builtin_amdgcn_cvt_pk_bf16_f32(x, y);
  return c.u;
#else
  return (unsigned int)f2bf(x) | ((unsigned int)f2bf(y) << 16);
#endif
}

// A fragment (16x32 bf16): lanes 0-15 row M=lane, halves = K{0..7,16..23};
// lanes 16-31 row M=lane-16, halves = K{8..15,24..31}. Two ds_load_b128.
static __device__ __forceinline__ v16bf load_a_frag(const unsigned short* s,
                                                    int row, int kb, int lane) {
  union { i4x q[2]; v16bf v; } u;
  const int r  = row + (lane & 15);
  const int k0 = kb + ((lane >> 4) << 3);
  u.q[0] = *(const i4x*)(s + r * LDA + k0);
  u.q[1] = *(const i4x*)(s + r * LDA + k0 + 16);
  return u.v;
}

// B fragment (32x16 bf16): lane holds column N=lane&15, halves = K{0..7,16..23}
// (lanes 0-15) / K{8..15,24..31} (lanes 16-31). LDS holds B transposed [N][K] -> two ds_load_b128.
static __device__ __forceinline__ v16bf load_b_frag(const unsigned short* s,
                                                    int col, int kb, int lane) {
  union { i4x q[2]; v16bf v; } u;
  const int c  = col + (lane & 15);
  const int k0 = kb + ((lane >> 4) << 3);
  u.q[0] = *(const i4x*)(s + c * LDB + k0);
  u.q[1] = *(const i4x*)(s + c * LDB + k0 + 16);
  return u.v;
}

// ---------------------------------------------------------------------------
// Kernel 1: gate_up GEMM (K=HID) + fused SwiGLU, output bf16 act [TOK][FFN]
// grid = (FFN/BN, TOK/BM, NEXP), block = 256
// ---------------------------------------------------------------------------
__global__ __launch_bounds__(256, 1)
void moe_gateup_swiglu(const float* __restrict__ hs,
                       const float* __restrict__ gup,
                       unsigned short* __restrict__ act) {
  __shared__ unsigned short sA [BM * LDA];
  __shared__ unsigned short sBg[BN * LDB];
  __shared__ unsigned short sBu[BN * LDB];

  const int tid   = threadIdx.x;
  const int lane  = tid & 31;
  const int wave  = tid >> 5;
  const int waveM = (wave >> 2) * 64;   // 2 waves in M
  const int waveN = (wave & 3) * 32;    // 4 waves in N

  const int e     = blockIdx.z;
  const int mTile = blockIdx.y * BM;
  const int nTile = blockIdx.x * BN;

  const float* Ag = hs  + (size_t)e * TOK * HID + (size_t)mTile * HID;
  const float* Bg = gup + (size_t)e * HID * (2 * FFN) + nTile;   // gate cols
  const float* Bu = Bg + FFN;                                    // up cols
  unsigned short* C = act + (size_t)e * TOK * FFN + (size_t)mTile * FFN + nTile;

  v8f accG[4][2], accU[4][2];
#pragma unroll
  for (int mi = 0; mi < 4; ++mi)
#pragma unroll
    for (int ni = 0; ni < 2; ++ni)
#pragma unroll
      for (int r = 0; r < 8; ++r) { accG[mi][ni][r] = 0.f; accU[mi][ni][r] = 0.f; }

  for (int k0 = 0; k0 < HID; k0 += BK) {
    // prefetch next-K tiles (emits global_prefetch_b8)
    if (k0 + BK < HID) {
      __builtin_prefetch(Ag + (size_t)(tid >> 1) * HID + (k0 + BK) + (tid & 1) * 32, 0, 0);
      const float* pb = Bg + (size_t)(k0 + BK + (tid >> 2)) * (2 * FFN) + (tid & 3) * 32;
      __builtin_prefetch(pb, 0, 0);
      __builtin_prefetch(pb + FFN, 0, 0);
    }

    // stage A: BM x BK fp32 -> bf16 row-major [BM][LDA]
#pragma unroll
    for (int t = 0; t < 8; ++t) {
      const int i   = tid + t * 256;
      const int row = i >> 4;
      const int c4  = (i & 15) << 2;
      f4x v = *(const f4x*)(Ag + (size_t)row * HID + k0 + c4);
      u2x p;
      p.x = pack2bf(v.x, v.y);
      p.y = pack2bf(v.z, v.w);
      *(u2x*)(sA + row * LDA + c4) = p;
    }

    // stage Bg/Bu: BK x BN fp32 -> bf16, transposed into [BN][LDB]
#pragma unroll
    for (int t = 0; t < 8; ++t) {
      const int i  = tid + t * 256;
      const int k  = i >> 5;
      const int n4 = (i & 31) << 2;
      f4x g = *(const f4x*)(Bg + (size_t)(k0 + k) * (2 * FFN) + n4);
      f4x u = *(const f4x*)(Bu + (size_t)(k0 + k) * (2 * FFN) + n4);
      unsigned int g01 = pack2bf(g.x, g.y), g23 = pack2bf(g.z, g.w);
      unsigned int u01 = pack2bf(u.x, u.y), u23 = pack2bf(u.z, u.w);
      sBg[(n4 + 0) * LDB + k] = (unsigned short)(g01 & 0xffffu);
      sBg[(n4 + 1) * LDB + k] = (unsigned short)(g01 >> 16);
      sBg[(n4 + 2) * LDB + k] = (unsigned short)(g23 & 0xffffu);
      sBg[(n4 + 3) * LDB + k] = (unsigned short)(g23 >> 16);
      sBu[(n4 + 0) * LDB + k] = (unsigned short)(u01 & 0xffffu);
      sBu[(n4 + 1) * LDB + k] = (unsigned short)(u01 >> 16);
      sBu[(n4 + 2) * LDB + k] = (unsigned short)(u23 & 0xffffu);
      sBu[(n4 + 3) * LDB + k] = (unsigned short)(u23 >> 16);
    }
    __syncthreads();

#pragma unroll
    for (int ks = 0; ks < 2; ++ks) {
      const int kb = ks * 32;
      v16bf af[4];
#pragma unroll
      for (int mi = 0; mi < 4; ++mi)
        af[mi] = load_a_frag(sA, waveM + mi * 16, kb, lane);
      v16bf bg[2], bu[2];
#pragma unroll
      for (int ni = 0; ni < 2; ++ni) {
        bg[ni] = load_b_frag(sBg, waveN + ni * 16, kb, lane);
        bu[ni] = load_b_frag(sBu, waveN + ni * 16, kb, lane);
      }
#pragma unroll
      for (int mi = 0; mi < 4; ++mi)
#pragma unroll
        for (int ni = 0; ni < 2; ++ni) {
          accG[mi][ni] = __builtin_amdgcn_wmma_f32_16x16x32_bf16(
              false, af[mi], false, bg[ni], (short)0, accG[mi][ni], false, false);
          accU[mi][ni] = __builtin_amdgcn_wmma_f32_16x16x32_bf16(
              false, af[mi], false, bu[ni], (short)0, accU[mi][ni], false, false);
        }
    }
    __syncthreads();
  }

  // epilogue: act = up * silu(gate), store bf16
  // C/D layout: VGPR r -> M = r + (lane>=16 ? 8 : 0), N = lane & 15
  const int rO = (lane >> 4) << 3;
  const int nO = lane & 15;
#pragma unroll
  for (int mi = 0; mi < 4; ++mi)
#pragma unroll
    for (int ni = 0; ni < 2; ++ni)
#pragma unroll
      for (int r = 0; r < 8; ++r) {
        const float g = accG[mi][ni][r];
        const float u = accU[mi][ni][r];
        const float s = u * (g / (1.0f + __expf(-g)));
        const int M = waveM + mi * 16 + rO + r;
        const int N = waveN + ni * 16 + nO;
        C[(size_t)M * FFN + N] = f2bf(s);
      }
}

// ---------------------------------------------------------------------------
// Kernel 2: down GEMM (K=FFN), A = bf16 act from ws, B = fp32 down_proj,
// output fp32. grid = (HID/BN, TOK/BM, NEXP), block = 256
// ---------------------------------------------------------------------------
__global__ __launch_bounds__(256, 1)
void moe_down(const unsigned short* __restrict__ act,
              const float* __restrict__ dwn,
              float* __restrict__ out) {
  __shared__ unsigned short sA[BM * LDA];
  __shared__ unsigned short sB[BN * LDB];

  const int tid   = threadIdx.x;
  const int lane  = tid & 31;
  const int wave  = tid >> 5;
  const int waveM = (wave >> 2) * 64;
  const int waveN = (wave & 3) * 32;

  const int e     = blockIdx.z;
  const int mTile = blockIdx.y * BM;
  const int nTile = blockIdx.x * BN;

  const unsigned short* Ag = act + (size_t)e * TOK * FFN + (size_t)mTile * FFN;
  const float* Bp = dwn + (size_t)e * FFN * HID + nTile;
  float* C = out + (size_t)e * TOK * HID + (size_t)mTile * HID + nTile;

  v8f acc[4][2];
#pragma unroll
  for (int mi = 0; mi < 4; ++mi)
#pragma unroll
    for (int ni = 0; ni < 2; ++ni)
#pragma unroll
      for (int r = 0; r < 8; ++r) acc[mi][ni][r] = 0.f;

  for (int k0 = 0; k0 < FFN; k0 += BK) {
    if (k0 + BK < FFN) {
      __builtin_prefetch(Ag + (size_t)(tid & 127) * FFN + (k0 + BK), 0, 0);
      __builtin_prefetch(Bp + (size_t)(k0 + BK + (tid >> 2)) * HID + (tid & 3) * 32, 0, 0);
    }

    // stage A: bf16 straight copy, [BM][LDA]
    // CDNA5 path: async global->LDS B128 copies (ASYNCcnt), else ds_store_b128
#pragma unroll
    for (int t = 0; t < 4; ++t) {
      const int i   = tid + t * 256;
      const int row = i >> 3;
      const int o8  = (i & 7) << 3;
#if __has_builtin(__builtin_amdgcn_global_load_async_to_lds_b128)
      __builtin_amdgcn_global_load_async_to_lds_b128(
          (gi4p)(Ag + (size_t)row * FFN + k0 + o8),
          (li4p)(sA + row * LDA + o8),
          0, 0);
#else
      *(i4x*)(sA + row * LDA + o8) =
          *(const i4x*)(Ag + (size_t)row * FFN + k0 + o8);
#endif
    }

    // stage B: BK x BN fp32 -> bf16 transposed [BN][LDB]
#pragma unroll
    for (int t = 0; t < 8; ++t) {
      const int i  = tid + t * 256;
      const int k  = i >> 5;
      const int n4 = (i & 31) << 2;
      f4x b = *(const f4x*)(Bp + (size_t)(k0 + k) * HID + n4);
      unsigned int b01 = pack2bf(b.x, b.y), b23 = pack2bf(b.z, b.w);
      sB[(n4 + 0) * LDB + k] = (unsigned short)(b01 & 0xffffu);
      sB[(n4 + 1) * LDB + k] = (unsigned short)(b01 >> 16);
      sB[(n4 + 2) * LDB + k] = (unsigned short)(b23 & 0xffffu);
      sB[(n4 + 3) * LDB + k] = (unsigned short)(b23 >> 16);
    }

#if __has_builtin(__builtin_amdgcn_global_load_async_to_lds_b128)
#if __has_builtin(__builtin_amdgcn_s_wait_asynccnt)
    __builtin_amdgcn_s_wait_asynccnt(0);
#else
    asm volatile("s_wait_asynccnt 0x0" ::: "memory");
#endif
#endif
    __syncthreads();

#pragma unroll
    for (int ks = 0; ks < 2; ++ks) {
      const int kb = ks * 32;
      v16bf af[4];
#pragma unroll
      for (int mi = 0; mi < 4; ++mi)
        af[mi] = load_a_frag(sA, waveM + mi * 16, kb, lane);
      v16bf bf[2];
#pragma unroll
      for (int ni = 0; ni < 2; ++ni)
        bf[ni] = load_b_frag(sB, waveN + ni * 16, kb, lane);
#pragma unroll
      for (int mi = 0; mi < 4; ++mi)
#pragma unroll
        for (int ni = 0; ni < 2; ++ni)
          acc[mi][ni] = __builtin_amdgcn_wmma_f32_16x16x32_bf16(
              false, af[mi], false, bf[ni], (short)0, acc[mi][ni], false, false);
    }
    __syncthreads();
  }

  const int rO = (lane >> 4) << 3;
  const int nO = lane & 15;
#pragma unroll
  for (int mi = 0; mi < 4; ++mi)
#pragma unroll
    for (int ni = 0; ni < 2; ++ni)
#pragma unroll
      for (int r = 0; r < 8; ++r) {
        const int M = waveM + mi * 16 + rO + r;
        const int N = waveN + ni * 16 + nO;
        C[(size_t)M * HID + N] = acc[mi][ni][r];
      }
}

// ---------------------------------------------------------------------------
extern "C" void kernel_launch(void* const* d_in, const int* in_sizes, int n_in,
                              void* d_out, int out_size, void* d_ws, size_t ws_size,
                              hipStream_t stream) {
  const float* hs  = (const float*)d_in[0];   // [NEXP*TOK, HID]
  const float* gup = (const float*)d_in[1];   // [NEXP, HID, 2*FFN]
  const float* dwn = (const float*)d_in[2];   // [NEXP, FFN, HID]
  float* out = (float*)d_out;                 // [NEXP*TOK, HID]
  unsigned short* actws = (unsigned short*)d_ws;  // bf16 act [NEXP, TOK, FFN] = 64 MB

  dim3 blk(256, 1, 1);
  dim3 g1(FFN / BN, TOK / BM, NEXP);   // 32 x 8 x 8
  dim3 g2(HID / BN, TOK / BM, NEXP);   // 16 x 8 x 8

  moe_gateup_swiglu<<<g1, blk, 0, stream>>>(hs, gup, actws);
  moe_down<<<g2, blk, 0, stream>>>(actws, dwn, out);
  (void)in_sizes; (void)n_in; (void)out_size; (void)ws_size;
}
